// SelfAttentionBlock_21414706938077
// MI455X (gfx1250) — matrix operand order repaired and checked by reading, not compiled, not run on hardware
//
#include <hip/hip_runtime.h>
#include <hip/hip_bf16.h>
#include <math.h>

// ---------------------------------------------------------------------------
// CDNA5 (gfx1250) wave32 WMMA self-attention block.
// qkv_proj -> flash attention (transposed S/acc, barrier-free, shfl-based
// fragment exchange) -> out_proj+residual (Wo transposed+swizzled in LDS).
// All GEMMs: v_wmma_f32_16x16x32_bf16 (f32 accumulate, bf16 operands).
// ---------------------------------------------------------------------------

typedef __attribute__((ext_vector_type(16))) __bf16       v16bf;
typedef __attribute__((ext_vector_type(8)))  __bf16       v8bf;
typedef __attribute__((ext_vector_type(8)))  float        v8f;
typedef __attribute__((ext_vector_type(4)))  unsigned int u32x4;

struct u32x8s { u32x4 lo, hi; };

#define BATCH 8
#define NTOK  4096           // 64*64 spatial positions per batch
#define CH    256
#define DQK   32             // CH/8
#define DV    128            // CH/2
#define ROWS  (BATCH * NTOK) // 32768 total rows
#define KBLK  64             // keys per flash-attention block

__device__ __forceinline__ v8f wmma_bf16(v16bf a, v16bf b, v8f c) {
    // D = A(16x32) * B(32x16) + C, f32 accumulate
    return __builtin_amdgcn_wmma_f32_16x16x32_bf16(
        /*neg_a=*/false, a, /*neg_b=*/false, b,
        /*c_mod=*/(short)0, c, /*reuse_a=*/false, /*reuse_b=*/false);
}

__device__ __forceinline__ v8f splat8(float v) {
    v8f r;
#pragma unroll
    for (int i = 0; i < 8; ++i) r[i] = v;
    return r;
}

// A-matrix 16x32 fragment from f32 row-major source (converted to bf16).
// Lanes 0-15: row M=lane, K = k0..k0+15;
// lanes 16-31: row M=lane-16, K = k0+16..k0+31.
__device__ __forceinline__ v16bf load_a_frag_f32(const float* __restrict__ src,
                                                 int ld, int row0, int k0, int lane) {
    const int m  = row0 + (lane & 15);
    const int kb = k0 + ((lane >> 4) << 4);
    const float* p = src + (size_t)m * ld + kb;
    v16bf a;
#pragma unroll
    for (int i = 0; i < 8; ++i) {
        a[i]     = (__bf16)p[i];
        a[8 + i] = (__bf16)p[8 + i];
    }
    return a;
}

// A-matrix 16x32 fragment from bf16 row-major source: two 16-B runs -> b128 x2.
__device__ __forceinline__ v16bf load_a_frag_bf16(const __bf16* __restrict__ src,
                                                  int ld, int row0, int k0, int lane) {
    const int m  = row0 + (lane & 15);
    const int kb = k0 + ((lane >> 4) << 4);
    const __bf16* p = src + (size_t)m * ld + kb;
    u32x8s t;
    t.lo = *(const u32x4*)p;
    t.hi = *(const u32x4*)(p + 8);
    return __builtin_bit_cast(v16bf, t);
}

// B-matrix 32x16 fragment, strided gather from f32 row-major B[k][n].
__device__ __forceinline__ v16bf load_b_frag_f32(const float* __restrict__ src,
                                                 int ld, int k0, int n0, int lane) {
    const int n  = n0 + (lane & 15);
    const int kb = k0 + ((lane >> 4) << 4);
    const float* p = src + (size_t)kb * ld + n;
    v16bf b;
#pragma unroll
    for (int i = 0; i < 16; ++i) b[i] = (__bf16)p[(size_t)i * ld];
    return b;
}

// B fragment where this lane's 16 K-elements are contiguous: b128 x2.
__device__ __forceinline__ v16bf load_b_frag_contig(const __bf16* __restrict__ p) {
    u32x8s t;
    t.lo = *(const u32x4*)p;
    t.hi = *(const u32x4*)(p + 8);
    return __builtin_bit_cast(v16bf, t);
}

// ---------------------------------------------------------------------------
// Kernel 1: q = xWq+bq, k = xWk+bk (bf16, row-major [row][d]);
//           v = xWv+bv stored TRANSPOSED bf16 [batch][DV][NTOK].
// Wv is staged transposed+swizzled in 64 KB LDS (conflict-free: 16 chunks
// XOR (n&15) fully covers the 512-B row stride). One wave per 16-row tile.
// ---------------------------------------------------------------------------
__global__ __launch_bounds__(256) void qkv_proj_kernel(
    const float* __restrict__ x,
    const float* __restrict__ Wq, const float* __restrict__ bq,
    const float* __restrict__ Wk, const float* __restrict__ bk,
    const float* __restrict__ Wv, const float* __restrict__ bv,
    unsigned short* __restrict__ qb_u, unsigned short* __restrict__ kb_u,
    unsigned short* __restrict__ vtb_u)
{
    __shared__ __bf16 wv_s[DV * CH];   // WvT[n(128)][k(256)] swizzled, 64 KB
    for (int i = threadIdx.x; i < CH * DV; i += 256) {
        const int k = i >> 7;          // Wv is [256][128] row-major: i = k*128+n
        const int n = i & 127;
        const int chunk = (k >> 4) ^ (n & 15);
        wv_s[n * CH + (chunk << 4) + (k & 15)] = (__bf16)Wv[i];
    }
    __syncthreads();

    __bf16* qb  = (__bf16*)qb_u;
    __bf16* kb  = (__bf16*)kb_u;
    __bf16* vtb = (__bf16*)vtb_u;

    const int lane = threadIdx.x & 31;
    const int tile = blockIdx.x * 8 + (threadIdx.x >> 5);
    const int row0 = tile << 4;

    // x tile as 8 A-fragments covering K = 0..255
    v16bf a[8];
#pragma unroll
    for (int kk = 0; kk < 8; ++kk)
        a[kk] = load_a_frag_f32(x, CH, row0, kk * 32, lane);

    const int colL  = lane & 15;
    const int rbase = (lane >> 4) << 3;
    const int khalf = (lane >> 4) << 4;

    // ---- Q and K: DQK=32 -> 2 N-tiles each (weights gathered via L1) ----
#pragma unroll
    for (int nt = 0; nt < 2; ++nt) {
        const int n0 = nt * 16;
        v8f cq = splat8(bq[n0 + colL]);
        v8f ck = splat8(bk[n0 + colL]);
#pragma unroll
        for (int kk = 0; kk < 8; ++kk) {
            cq = wmma_bf16(a[kk], load_b_frag_f32(Wq, DQK, kk * 32, n0, lane), cq);
            ck = wmma_bf16(a[kk], load_b_frag_f32(Wk, DQK, kk * 32, n0, lane), ck);
        }
#pragma unroll
        for (int r = 0; r < 8; ++r) {
            const int row = row0 + rbase + r;
            qb[(size_t)row * DQK + n0 + colL] = (__bf16)cq[r];
            kb[(size_t)row * DQK + n0 + colL] = (__bf16)ck[r];
        }
    }

    // ---- V: DV=128 -> 8 N-tiles, B-frags from LDS, transposed b128 store ----
    const int bidx = row0 >> 12;    // row0 / NTOK
    const int nloc = row0 & (NTOK - 1);
#pragma unroll
    for (int nt = 0; nt < 8; ++nt) {
        const int n0 = nt * 16;
        const int n  = n0 + colL;
        v8f cv = splat8(bv[n]);
#pragma unroll
        for (int kk = 0; kk < 8; ++kk) {
            const int kbx = kk * 32 + khalf;           // multiple of 16
            const int chunk = (kbx >> 4) ^ (n & 15);
            cv = wmma_bf16(a[kk],
                           load_b_frag_contig(wv_s + n * CH + (chunk << 4)), cv);
        }
        // vT[b][vdim][token]: 8 consecutive tokens per lane -> one b128 store
        v8bf pk;
#pragma unroll
        for (int r = 0; r < 8; ++r) pk[r] = (__bf16)cv[r];
        *(v8bf*)(vtb + ((size_t)(bidx * DV + n) * NTOK + nloc + rbase)) = pk;
    }
}

// ---------------------------------------------------------------------------
// Kernel 2: flash attention, fully transposed & barrier-free.
//   S^T[64k x 16q] = K_blk x Q^T    (4 WMMA)
//   acc^T[128v x 16q] += V^T x P^T  (16 WMMA)
// In the transposed layout each lane owns one query's scores: softmax max/sum
// are local reductions + ONE shfl_xor(16) each; alpha/m/l are per-lane
// scalars; the S^T C-tile -> P^T B-fragment relayout is a single xor-16
// lane-pair exchange (no LDS, no __syncthreads anywhere in the loop).
// ---------------------------------------------------------------------------
__global__ __launch_bounds__(256) void attn_kernel(
    const unsigned short* __restrict__ qb_u, const unsigned short* __restrict__ kb_u,
    const unsigned short* __restrict__ vtb_u, float* __restrict__ ao)
{
    const __bf16* qb  = (const __bf16*)qb_u;
    const __bf16* kp0 = (const __bf16*)kb_u;
    const __bf16* vtb = (const __bf16*)vtb_u;

    const int lane = threadIdx.x & 31;
    const int tile = blockIdx.x * 8 + (threadIdx.x >> 5);
    const int b    = tile >> 8;           // tile / (NTOK/16)
    const int qt   = tile & 255;

    const __bf16* qp = qb  + (size_t)(b * NTOK + qt * 16) * DQK;
    const __bf16* kp = kp0 + (size_t)b * NTOK * DQK;
    const __bf16* vp = vtb + (size_t)b * DV * NTOK;

    const int  colL  = lane & 15;
    const int  khalf = (lane >> 4) << 4;
    const int  mrow  = (lane >> 4) << 3;
    const bool isLo  = (lane < 16);

    // Q^T B-fragment (B(kdim, query)), loaded once: contiguous per lane
    const v16bf qtf = load_b_frag_contig(qp + (size_t)colL * DQK + khalf);

    v8f acc[8];                            // acc^T: M=vdim tiles, N=query
#pragma unroll
    for (int t = 0; t < 8; ++t) acc[t] = v8f{};
    float m = -3.0e38f, l = 0.0f;          // per-query (per-lane) stats

    for (int key0 = 0; key0 < NTOK; key0 += KBLK) {
        // prefetch next key block (K rows + leading V rows)
        if (key0 + KBLK < NTOK) {
            __builtin_prefetch(kp + (size_t)(key0 + KBLK + colL) * DQK, 0, 3);
            __builtin_prefetch(vp + (size_t)lane * NTOK + key0 + KBLK, 0, 3);
        }

        // S^T tiles: A = K rows (keys), B = Q^T
        v8f sT[4];
#pragma unroll
        for (int st = 0; st < 4; ++st) {
            v16bf ka = load_a_frag_bf16(kp, DQK, key0 + st * 16, 0, lane);
            v8f z = v8f{};
            sT[st] = wmma_bf16(ka, qtf, z);
        }

        // online softmax: local reduce over this lane's 32 keys + 1 shfl each
        float mloc = sT[0][0];
#pragma unroll
        for (int st = 0; st < 4; ++st)
#pragma unroll
            for (int r = 0; r < 8; ++r) mloc = fmaxf(mloc, sT[st][r]);
        mloc = fmaxf(mloc, __shfl_xor(mloc, 16));
        const float mn = fmaxf(m, mloc);
        const float al = __expf(m - mn);
        float ssum = 0.0f;
#pragma unroll
        for (int st = 0; st < 4; ++st)
#pragma unroll
            for (int r = 0; r < 8; ++r) {
                const float pv = __expf(sT[st][r] - mn);
                sT[st][r] = pv;
                ssum += pv;
            }
        ssum += __shfl_xor(ssum, 16);
        l = l * al + ssum;
        m = mn;

        // P^T B-fragments: keys striped across lane pairs -> xor-16 exchange.
        // frag0 covers keys key0..key0+31 (tiles 0,1), frag1 keys +32..+63.
        v16bf pf0, pf1;
#pragma unroll
        for (int r = 0; r < 8; ++r) {
            const float x0 = __shfl_xor(sT[0][r], 16);
            const float x1 = __shfl_xor(sT[1][r], 16);
            const float x2 = __shfl_xor(sT[2][r], 16);
            const float x3 = __shfl_xor(sT[3][r], 16);
            pf0[r]     = (__bf16)(isLo ? sT[0][r] : x1);
            pf0[8 + r] = (__bf16)(isLo ? x0 : sT[1][r]);
            pf1[r]     = (__bf16)(isLo ? sT[2][r] : x3);
            pf1[8 + r] = (__bf16)(isLo ? x2 : sT[3][r]);
        }

        // acc^T = acc^T*al + V^T_blk * P^T (al is a per-lane scalar)
#pragma unroll
        for (int t = 0; t < 8; ++t) {
#pragma unroll
            for (int r = 0; r < 8; ++r) acc[t][r] *= al;
            v16bf va0 = load_a_frag_bf16(vp, NTOK, t * 16, key0,      lane);
            v16bf va1 = load_a_frag_bf16(vp, NTOK, t * 16, key0 + 32, lane);
            acc[t] = wmma_bf16(va0, pf0, acc[t]);
            acc[t] = wmma_bf16(va1, pf1, acc[t]);
        }
    }

    // epilogue: row = this lane's query, vdim contiguous -> float4 stores
    const float inv = 1.0f / l;
    float* op = ao + (size_t)(b * NTOK + qt * 16 + colL) * DV;
#pragma unroll
    for (int t = 0; t < 8; ++t) {
        float4* dst = (float4*)(op + t * 16 + mrow);
        dst[0] = make_float4(acc[t][0] * inv, acc[t][1] * inv,
                             acc[t][2] * inv, acc[t][3] * inv);
        dst[1] = make_float4(acc[t][4] * inv, acc[t][5] * inv,
                             acc[t][6] * inv, acc[t][7] * inv);
    }
}

// ---------------------------------------------------------------------------
// Kernel 3: out = attn_out @ Wo + bo + x.
// Wo staged TRANSPOSED (WoT[n][k]) bf16 in 64 KB LDS; 16-element K-chunks
// XOR-swizzled by (n&7) so each lane's B-fragment is 2x ds_load_b128.
// ---------------------------------------------------------------------------
__global__ __launch_bounds__(256) void out_proj_kernel(
    const float* __restrict__ ao, const float* __restrict__ Wo,
    const float* __restrict__ bo, const float* __restrict__ x,
    float* __restrict__ out)
{
    __shared__ __bf16 wo_s[CH * DV];   // 256*128*2B = 64 KB, swizzled WoT
    for (int i = threadIdx.x; i < DV * CH; i += 256) {
        const int k = i >> 8;          // Wo is [128][256] row-major: i = k*256+n
        const int n = i & 255;
        const int chunk = (k >> 4) ^ (n & 7);
        wo_s[n * DV + (chunk << 4) + (k & 15)] = (__bf16)Wo[i];
    }
    __syncthreads();

    const int lane = threadIdx.x & 31;
    const int tile = blockIdx.x * 8 + (threadIdx.x >> 5);
    const int row0 = tile << 4;

    v16bf a[4];
#pragma unroll
    for (int kk = 0; kk < 4; ++kk)
        a[kk] = load_a_frag_f32(ao, DV, row0, kk * 32, lane);

    const int colL  = lane & 15;
    const int rbase = (lane >> 4) << 3;
    const int khalf = (lane >> 4) << 4;

#pragma unroll
    for (int nt = 0; nt < 16; ++nt) {
        const int n0 = nt * 16;
        const int n  = n0 + colL;
        v8f c = splat8(bo[n]);
#pragma unroll
        for (int kk = 0; kk < 4; ++kk) {
            const int kb = kk * 32 + khalf;            // multiple of 16
            const int chunk = (kb >> 4) ^ (n & 7);
            c = wmma_bf16(a[kk],
                          load_b_frag_contig(wo_s + n * DV + (chunk << 4)), c);
        }
#pragma unroll
        for (int r = 0; r < 8; ++r) {
            const size_t idx = (size_t)(row0 + rbase + r) * CH + n0 + colL;
            out[idx] = c[r] + x[idx];
        }
    }
}

// ---------------------------------------------------------------------------
extern "C" void kernel_launch(void* const* d_in, const int* in_sizes, int n_in,
                              void* d_out, int out_size, void* d_ws, size_t ws_size,
                              hipStream_t stream) {
    (void)in_sizes; (void)n_in; (void)out_size; (void)ws_size;

    const float* x  = (const float*)d_in[0];
    const float* Wq = (const float*)d_in[1];
    const float* bq = (const float*)d_in[2];
    const float* Wk = (const float*)d_in[3];
    const float* bk = (const float*)d_in[4];
    const float* Wv = (const float*)d_in[5];
    const float* bv = (const float*)d_in[6];
    const float* Wo = (const float*)d_in[7];
    const float* bo = (const float*)d_in[8];
    float* out = (float*)d_out;

    // workspace layout (28 MB): qb 2MB | kb 2MB | vT 8MB | attn_out 16MB
    unsigned short* qb  = (unsigned short*)d_ws;
    unsigned short* kb  = qb + (size_t)ROWS * DQK;
    unsigned short* vtb = kb + (size_t)ROWS * DQK;
    float*          ao  = (float*)(vtb + (size_t)BATCH * DV * NTOK);

    const int blocks = ROWS / 16 / 8;   // 2048 wave-tiles / 8 waves per block
    qkv_proj_kernel<<<blocks, 256, 0, stream>>>(x, Wq, bq, Wk, bk, Wv, bv,
                                                qb, kb, vtb);
    attn_kernel<<<blocks, 256, 0, stream>>>(qb, kb, vtb, ao);
    out_proj_kernel<<<blocks, 256, 0, stream>>>(ao, Wo, bo, x, out);
}
